// GRU_12128987644406
// MI455X (gfx1250) — matrix operand order repaired and checked by reading, not compiled, not run on hardware
//
#include <hip/hip_runtime.h>
#include <hip/hip_bf16.h>

typedef __attribute__((ext_vector_type(16))) _Float16 v16h;
typedef __attribute__((ext_vector_type(8)))  _Float16 v8h;
typedef __attribute__((ext_vector_type(8)))  float    v8f;

#define GRU_B 128
#define GRU_T 1024
#define GRU_D 64
#define GRU_H 256
#define GRU_G 768   // 3*H

__device__ __forceinline__ float sigf(float x) {
    return 1.0f / (1.0f + __expf(-x));
}

// ---------------------------------------------------------------------------
// f32 -> f16 weight conversion
// ---------------------------------------------------------------------------
__global__ void cvt_f32_f16(const float* __restrict__ src,
                            _Float16* __restrict__ dst, int n) {
    int i = blockIdx.x * blockDim.x + threadIdx.x;
    if (i < n) dst[i] = (_Float16)src[i];
}

// ---------------------------------------------------------------------------
// Input-gate GEMM:  out[M, 768] = A[M, K] * W[768, K]^T + bias
// A is f32 (layer 0, K=64) or f16 (layer 1, K=256). One block = 256 threads
// (8 waves); block tile = 16 rows x 128 gates (each wave one 16x16 tile strip).
// ---------------------------------------------------------------------------
template <int K, bool A_F32>
__global__ __launch_bounds__(256) void gates_gemm(
    const void* __restrict__ A_, const _Float16* __restrict__ W,
    const float* __restrict__ bias, float* __restrict__ out) {
    __shared__ __align__(16) _Float16 atile[16 * K];

    const int tid = threadIdx.x;
    const long row0 = (long)blockIdx.y * 16;

    // Cooperative load of the 16xK A tile into LDS (converted to f16).
    if (A_F32) {
        const float* A = (const float*)A_;
        for (int i = tid; i < 16 * K; i += 256)
            atile[i] = (_Float16)A[(row0 + (i / K)) * K + (i % K)];
    } else {
        const _Float16* A = (const _Float16*)A_;
        for (int i = tid; i < 16 * K; i += 256)
            atile[i] = A[(row0 + (i / K)) * K + (i % K)];
    }
    __syncthreads();

    const int wid  = tid >> 5;
    const int lane = tid & 31;
    const int g0   = blockIdx.x * 128 + wid * 16;   // gate column base of tile
    const int am   = lane & 15;                     // A row within tile
    const int akhi = (lane >> 4) * 8;               // A K sub-offset
    const int bn   = lane & 15;                     // B column within tile
    const int bkb  = (lane >> 4) * 8;               // B K sub-offset

    v8f acc = {};
#pragma unroll
    for (int kc = 0; kc < K / 32; ++kc) {
        const _Float16* ap = &atile[am * K + kc * 32 + akhi];
        v8h alo = *(const v8h*)ap;
        v8h ahi = *(const v8h*)(ap + 16);
        v16h a;
#pragma unroll
        for (int i = 0; i < 8; ++i) { a[i] = alo[i]; a[8 + i] = ahi[i]; }

        const _Float16* bp = &W[(long)(g0 + bn) * K + kc * 32 + bkb];
        v8h blo = *(const v8h*)bp;
        v8h bhi = *(const v8h*)(bp + 16);
        v16h b;
#pragma unroll
        for (int i = 0; i < 8; ++i) { b[i] = blo[i]; b[8 + i] = bhi[i]; }

        acc = __builtin_amdgcn_wmma_f32_16x16x32_f16(
            false, a, false, b, (short)0, acc, false, false);
    }

    const float bv = bias[g0 + bn];
    const int mb = (lane >> 4) * 8;
#pragma unroll
    for (int i = 0; i < 8; ++i)
        out[(row0 + mb + i) * GRU_G + g0 + bn] = acc[i] + bv;
}

// ---------------------------------------------------------------------------
// Persistent GRU recurrence. One block per 16-row batch slab, loops over all
// T steps. h kept in LDS (f32 master + f16 copy for WMMA A operands).
// Wave w owns H-slice [32w, 32w+32): 6 tiles (r,z,n x 2 j-tiles) in registers,
// so the gate nonlinearity needs no cross-wave staging.
// ---------------------------------------------------------------------------
__global__ __launch_bounds__(256) void gru_recur(
    const float* __restrict__ xg,      // [B, T, 768] precomputed input gates
    const _Float16* __restrict__ Whh,  // [768, 256] f16
    const float* __restrict__ b_hh,    // [768]
    _Float16* __restrict__ seq_out,    // [B, T, 256] f16 (if store_seq)
    float* __restrict__ h_last,        // [B, 256] f32 (if write_last)
    int store_seq, int write_last) {
    __shared__ __align__(16) _Float16 hc16[16 * GRU_H];
    __shared__ float hc32[16 * GRU_H];

    const int tid  = threadIdx.x;
    const int wid  = tid >> 5;
    const int lane = tid & 31;
    const int b0   = blockIdx.x * 16;

    // h = 0
    for (int i = tid; i < 16 * GRU_H; i += 256) {
        hc32[i] = 0.0f;
        hc16[i] = (_Float16)0.0f;
    }
    __syncthreads();

    const int jt0  = wid * 2;            // this wave's two 16-wide j tiles
    const int am   = lane & 15;
    const int akhi = (lane >> 4) * 8;
    const int bn   = lane & 15;
    const int bkb  = (lane >> 4) * 8;
    const int mb   = (lane >> 4) * 8;
    const int jl   = lane & 15;

    // Per-lane recurrent-bias values (fixed columns), hoisted out of t-loop.
    float bh[6];
#pragma unroll
    for (int u = 0; u < 6; ++u) {
        const int gi = u >> 1, p = u & 1;
        bh[u] = b_hh[gi * GRU_H + (jt0 + p) * 16 + jl];
    }
    // Per-lane xg row bases (8 batch rows this lane touches).
    long xrow[8];
#pragma unroll
    for (int i = 0; i < 8; ++i)
        xrow[i] = (long)(b0 + mb + i) * GRU_T * GRU_G;

    for (int t = 0; t < GRU_T; ++t) {
        // ---- hidden GEMM phase: hg = h @ Whh^T (6 16x16 tiles per wave) ---
        v8f acc[6] = {{}, {}, {}, {}, {}, {}};
#pragma unroll
        for (int kc = 0; kc < GRU_H / 32; ++kc) {
            const _Float16* ap = &hc16[am * GRU_H + kc * 32 + akhi];
            v8h alo = *(const v8h*)ap;
            v8h ahi = *(const v8h*)(ap + 16);
            v16h a;
#pragma unroll
            for (int i = 0; i < 8; ++i) { a[i] = alo[i]; a[8 + i] = ahi[i]; }
#pragma unroll
            for (int u = 0; u < 6; ++u) {
                const int gi = u >> 1, p = u & 1;
                const int g0 = gi * GRU_H + (jt0 + p) * 16;
                const _Float16* bp =
                    &Whh[(long)(g0 + bn) * GRU_H + kc * 32 + bkb];
                v8h blo = *(const v8h*)bp;
                v8h bhi = *(const v8h*)(bp + 16);
                v16h b;
#pragma unroll
                for (int i = 0; i < 8; ++i) { b[i] = blo[i]; b[8 + i] = bhi[i]; }
                acc[u] = __builtin_amdgcn_wmma_f32_16x16x32_f16(
                    false, a, false, b, (short)0, acc[u], false, false);
            }
        }
        __syncthreads();   // all waves finished reading hc16

        // ---- gate nonlinearity + state update (in registers) --------------
        if (t + 1 < GRU_T)  // prefetch next step's gate rows toward L2/L0
            __builtin_prefetch(&xg[xrow[0] + (long)(t + 1) * GRU_G], 0, 1);

#pragma unroll
        for (int p = 0; p < 2; ++p) {
            const int j = (jt0 + p) * 16 + jl;
#pragma unroll
            for (int i = 0; i < 8; ++i) {
                const int m = mb + i;
                const float* xr = &xg[xrow[i] + (long)t * GRU_G];
                float r  = sigf(xr[j]             + acc[0 + p][i] + bh[0 + p]);
                float z  = sigf(xr[GRU_H + j]     + acc[2 + p][i] + bh[2 + p]);
                float nn = tanhf(xr[2 * GRU_H + j] +
                                 r * (acc[4 + p][i] + bh[4 + p]));
                float h    = hc32[m * GRU_H + j];
                float hnew = (1.0f - z) * nn + z * h;
                hc32[m * GRU_H + j] = hnew;
                hc16[m * GRU_H + j] = (_Float16)hnew;
                if (store_seq)
                    seq_out[((long)(b0 + m) * GRU_T + t) * GRU_H + j] =
                        (_Float16)hnew;
            }
        }
        __syncthreads();   // hc16 updated before next step's GEMM
    }

    if (write_last) {
        for (int i = tid; i < 16 * GRU_H; i += 256)
            h_last[(long)(b0 + i / GRU_H) * GRU_H + (i % GRU_H)] = hc32[i];
    }
}

// ---------------------------------------------------------------------------
// FC on last timestep: out[b] = h2[b,:] . fc_w + fc_b   (OUT = 1)
// ---------------------------------------------------------------------------
__global__ void fc_kernel(const float* __restrict__ h,
                          const float* __restrict__ w,
                          const float* __restrict__ b,
                          float* __restrict__ out) {
    int bi = blockIdx.x * blockDim.x + threadIdx.x;
    if (bi < GRU_B) {
        float s = 0.0f;
#pragma unroll 8
        for (int j = 0; j < GRU_H; ++j) s += h[bi * GRU_H + j] * w[j];
        out[bi] = s + b[0];
    }
}

// ---------------------------------------------------------------------------
extern "C" void kernel_launch(void* const* d_in, const int* in_sizes, int n_in,
                              void* d_out, int out_size, void* d_ws,
                              size_t ws_size, hipStream_t stream) {
    const float* x     = (const float*)d_in[0];
    const float* W_ih0 = (const float*)d_in[1];
    const float* W_hh0 = (const float*)d_in[2];
    const float* b_ih0 = (const float*)d_in[3];
    const float* b_hh0 = (const float*)d_in[4];
    const float* W_ih1 = (const float*)d_in[5];
    const float* W_hh1 = (const float*)d_in[6];
    const float* b_ih1 = (const float*)d_in[7];
    const float* b_hh1 = (const float*)d_in[8];
    const float* fc_w  = (const float*)d_in[9];
    const float* fc_b  = (const float*)d_in[10];
    float* out = (float*)d_out;

    // Workspace layout
    char* ws = (char*)d_ws;
    _Float16* Wih0h = (_Float16*)ws;                      // 768*64
    _Float16* Whh0h = Wih0h + GRU_G * GRU_D;              // 768*256
    _Float16* Wih1h = Whh0h + GRU_G * GRU_H;              // 768*256
    _Float16* Whh1h = Wih1h + GRU_G * GRU_H;              // 768*256
    float*    xg    = (float*)(ws + (2ull << 20));        // [B*T, 768] f32
    const size_t xg_bytes = (size_t)GRU_B * GRU_T * GRU_G * sizeof(float);
    _Float16* h1seq = (_Float16*)((char*)xg + xg_bytes);  // [B*T, 256] f16
    const size_t h1_bytes = (size_t)GRU_B * GRU_T * GRU_H * sizeof(_Float16);
    float* h2last = (float*)((char*)h1seq + h1_bytes);    // [B, 256] f32

    // 1) weights to f16
    cvt_f32_f16<<<(GRU_G * GRU_D + 255) / 256, 256, 0, stream>>>(W_ih0, Wih0h,
                                                                 GRU_G * GRU_D);
    cvt_f32_f16<<<(GRU_G * GRU_H + 255) / 256, 256, 0, stream>>>(W_hh0, Whh0h,
                                                                 GRU_G * GRU_H);
    cvt_f32_f16<<<(GRU_G * GRU_H + 255) / 256, 256, 0, stream>>>(W_ih1, Wih1h,
                                                                 GRU_G * GRU_H);
    cvt_f32_f16<<<(GRU_G * GRU_H + 255) / 256, 256, 0, stream>>>(W_hh1, Whh1h,
                                                                 GRU_G * GRU_H);

    const int Mtiles = (GRU_B * GRU_T) / 16;              // 8192
    // 2) layer-0 input gates: xg = x @ W_ih0^T + b_ih0
    gates_gemm<GRU_D, true><<<dim3(GRU_G / 128, Mtiles), 256, 0, stream>>>(
        (const void*)x, Wih0h, b_ih0, xg);
    // 3) layer-0 recurrence -> h1seq (f16)
    gru_recur<<<GRU_B / 16, 256, 0, stream>>>(xg, Whh0h, b_hh0, h1seq, h2last,
                                              1, 0);
    // 4) layer-1 input gates: xg = h1seq @ W_ih1^T + b_ih1
    gates_gemm<GRU_H, false><<<dim3(GRU_G / 128, Mtiles), 256, 0, stream>>>(
        (const void*)h1seq, Wih1h, b_ih1, xg);
    // 5) layer-1 recurrence -> final hidden state only
    gru_recur<<<GRU_B / 16, 256, 0, stream>>>(xg, Whh1h, b_hh1, h1seq, h2last,
                                              0, 1);
    // 6) FC on last timestep
    fc_kernel<<<1, 128, 0, stream>>>(h2last, fc_w, fc_b, out);
}
